// TernaryLinearTorch_20976620274241
// MI455X (gfx1250) — matrix operand order repaired and checked by reading, not compiled, not run on hardware
//
#include <hip/hip_runtime.h>

typedef __attribute__((ext_vector_type(16))) _Float16 v16h;
typedef __attribute__((ext_vector_type(8)))  _Float16 v8h;
typedef __attribute__((ext_vector_type(4)))  _Float16 v4h;
typedef __attribute__((ext_vector_type(8)))  float    v8f;

#define N_ROWS 8192
#define K_DIM  4096
#define O_DIM  4096
#define BLK_M  128
#define BLK_N  128
#define BLK_K  32
#define LDS_STRIDE 40   // 32 halves + 8 pad -> conflict-free b128 frag loads
#define KT (K_DIM / BLK_K)

// ---------------- threshold reduction (deterministic, two-pass) ----------------

__global__ __launch_bounds__(256) void absmean_partial(const float* __restrict__ w,
                                                       float* __restrict__ partial,
                                                       unsigned int n)
{
    __shared__ float red[256];
    const int tid = threadIdx.x;
    const unsigned int stride = gridDim.x * 256u;
    float s = 0.0f;
    for (unsigned int i = blockIdx.x * 256u + tid; i < n; i += stride)
        s += fabsf(w[i]);
    red[tid] = s;
    __syncthreads();
    #pragma unroll
    for (int off = 128; off > 0; off >>= 1) {
        if (tid < off) red[tid] += red[tid + off];
        __syncthreads();
    }
    if (tid == 0) partial[blockIdx.x] = red[0];
}

__global__ __launch_bounds__(256) void absmean_final(const float* __restrict__ partial,
                                                     float* __restrict__ th,
                                                     int nparts, float inv_n)
{
    __shared__ float red[256];
    const int tid = threadIdx.x;
    float s = 0.0f;
    for (int i = tid; i < nparts; i += 256) s += partial[i];
    red[tid] = s;
    __syncthreads();
    #pragma unroll
    for (int off = 128; off > 0; off >>= 1) {
        if (tid < off) red[tid] += red[tid + off];
        __syncthreads();
    }
    if (tid == 0) th[0] = 0.5f * red[0] * inv_n;
}

// ---------------- WMMA GEMM ----------------

__device__ __forceinline__ _Float16 tq(float v, float th)
{
    float r = (v > th) ? 1.0f : ((v < -th) ? -1.0f : 0.0f);
    return (_Float16)r;
}

__device__ __forceinline__ void load_tile(const float* __restrict__ x,
                                          const float* __restrict__ w,
                                          int mBase, int nBase, int k0, int tid,
                                          float4 aReg[4], float4 bReg[4])
{
    #pragma unroll
    for (int i = 0; i < 4; ++i) {
        int f   = i * 256 + tid;      // 0..1023 float4 slots
        int row = f >> 3;             // 8 float4 per 32-wide row
        int c4  = f & 7;
        aReg[i] = *(const float4*)(x + (size_t)(mBase + row) * K_DIM + k0 + c4 * 4);
        bReg[i] = *(const float4*)(w + (size_t)(nBase + row) * K_DIM + k0 + c4 * 4);
    }
}

__device__ __forceinline__ void store_tile(_Float16* __restrict__ Ad,
                                           _Float16* __restrict__ Bd,
                                           int tid, float th,
                                           const float4 aReg[4], const float4 bReg[4])
{
    #pragma unroll
    for (int i = 0; i < 4; ++i) {
        int f   = i * 256 + tid;
        int row = f >> 3;
        int c4  = f & 7;
        v4h ah;
        ah.x = (_Float16)aReg[i].x; ah.y = (_Float16)aReg[i].y;
        ah.z = (_Float16)aReg[i].z; ah.w = (_Float16)aReg[i].w;
        *(v4h*)(Ad + row * LDS_STRIDE + c4 * 4) = ah;
        v4h bh;
        bh.x = tq(bReg[i].x, th); bh.y = tq(bReg[i].y, th);
        bh.z = tq(bReg[i].z, th); bh.w = tq(bReg[i].w, th);
        *(v4h*)(Bd + row * LDS_STRIDE + c4 * 4) = bh;
    }
}

__global__ __launch_bounds__(256) void ternary_gemm(const float* __restrict__ x,
                                                    const float* __restrict__ w,
                                                    const float* __restrict__ bias,
                                                    const float* __restrict__ gamma,
                                                    const float* __restrict__ thresh,
                                                    float* __restrict__ out)
{
    __shared__ __align__(16) _Float16 Ash[2][BLK_M * LDS_STRIDE];
    __shared__ __align__(16) _Float16 Bsh[2][BLK_N * LDS_STRIDE];

    const int tid   = threadIdx.x;
    const int lane  = tid & 31;
    const int wave  = tid >> 5;
    const int wm    = wave >> 1;         // 0..3 -> 32-row slab
    const int wn    = wave & 1;          // 0..1 -> 64-col slab
    const int mBase = blockIdx.y * BLK_M;
    const int nBase = blockIdx.x * BLK_N;

    const float th = thresh[0];
    const float g  = gamma[0];

    const int lhalf = lane >> 4;         // 0/1 selects K sub-chunk per ISA layout
    const int l16   = lane & 15;

    v8f acc[2][4];
    const v8f vzero = {0.f, 0.f, 0.f, 0.f, 0.f, 0.f, 0.f, 0.f};
    #pragma unroll
    for (int mt = 0; mt < 2; ++mt)
        #pragma unroll
        for (int nt = 0; nt < 4; ++nt)
            acc[mt][nt] = vzero;

    float4 aReg[4], bReg[4];

    // prologue: stage tile 0
    load_tile(x, w, mBase, nBase, 0, tid, aReg, bReg);
    store_tile(Ash[0], Bsh[0], tid, th, aReg, bReg);
    __syncthreads();

    for (int it = 0; it < KT; ++it) {
        const int  cur     = it & 1;
        const bool hasNext = (it + 1 < KT);

        if (hasNext)
            load_tile(x, w, mBase, nBase, (it + 1) * BLK_K, tid, aReg, bReg);

        const _Float16* Ab = Ash[cur];
        const _Float16* Bb = Bsh[cur];

        // A fragments: lane holds row (l16), K = 8*lhalf..+7 and 16+8*lhalf..+7
        v16h afrag[2];
        #pragma unroll
        for (int mt = 0; mt < 2; ++mt) {
            int row = wm * 32 + mt * 16 + l16;
            v8h lo = *(const v8h*)(Ab + row * LDS_STRIDE + lhalf * 8);
            v8h hi = *(const v8h*)(Ab + row * LDS_STRIDE + lhalf * 8 + 16);
            afrag[mt] = __builtin_shufflevector(lo, hi, 0,1,2,3,4,5,6,7,8,9,10,11,12,13,14,15);
        }
        // B fragments: lane holds column (l16), K = 8*lhalf..+7 and 16+8*lhalf..+7
        v16h bfrag[4];
        #pragma unroll
        for (int nt = 0; nt < 4; ++nt) {
            int col = wn * 64 + nt * 16 + l16;
            v8h lo = *(const v8h*)(Bb + col * LDS_STRIDE + lhalf * 8);
            v8h hi = *(const v8h*)(Bb + col * LDS_STRIDE + lhalf * 8 + 16);
            bfrag[nt] = __builtin_shufflevector(lo, hi, 0,1,2,3,4,5,6,7,8,9,10,11,12,13,14,15);
        }

        #pragma unroll
        for (int mt = 0; mt < 2; ++mt)
            #pragma unroll
            for (int nt = 0; nt < 4; ++nt)
                acc[mt][nt] = __builtin_amdgcn_wmma_f32_16x16x32_f16(
                    false, afrag[mt], false, bfrag[nt],
                    (short)0, acc[mt][nt], false, false);

        if (hasNext) {
            store_tile(Ash[cur ^ 1], Bsh[cur ^ 1], tid, th, aReg, bReg);
            __syncthreads();
        }
    }

    // epilogue: D layout — lane holds column l16, rows 8*lhalf + r
    const int lrow = lhalf * 8;
    #pragma unroll
    for (int mt = 0; mt < 2; ++mt) {
        #pragma unroll
        for (int nt = 0; nt < 4; ++nt) {
            const int gm = mBase + wm * 32 + mt * 16 + lrow;
            const int gn = nBase + wn * 64 + nt * 16 + l16;
            const float b = bias[gn];
            #pragma unroll
            for (int r = 0; r < 8; ++r)
                out[(size_t)(gm + r) * O_DIM + gn] = (acc[mt][nt][r] + b) * g;
        }
    }
}

// ---------------- launch ----------------

extern "C" void kernel_launch(void* const* d_in, const int* in_sizes, int n_in,
                              void* d_out, int out_size, void* d_ws, size_t ws_size,
                              hipStream_t stream)
{
    const float* x      = (const float*)d_in[0];
    const float* weight = (const float*)d_in[1];
    const float* bias   = (const float*)d_in[2];
    const float* gamma  = (const float*)d_in[3];
    float*       out    = (float*)d_out;

    float* ws      = (float*)d_ws;
    float* partial = ws;          // 512 floats
    float* th      = ws + 512;    // 1 float

    const unsigned int wn = (unsigned int)O_DIM * (unsigned int)K_DIM;

    absmean_partial<<<512, 256, 0, stream>>>(weight, partial, wn);
    absmean_final<<<1, 256, 0, stream>>>(partial, th, 512,
                                         1.0f / ((float)O_DIM * (float)K_DIM));

    dim3 grid(O_DIM / BLK_N, N_ROWS / BLK_M);
    ternary_gemm<<<grid, 256, 0, stream>>>(x, weight, bias, gamma, th, out);
}
